// CoKT_15831249453388
// MI455X (gfx1250) — matrix operand order repaired and checked by reading, not compiled, not run on hardware
//
#include <hip/hip_runtime.h>
#include <math.h>

// ---------------------------------------------------------------------------
// CoKT pipeline for MI455X (gfx1250, wave32, WMMA f16 16x16x32, f32 accum)
// ---------------------------------------------------------------------------

typedef _Float16 half_t;
typedef __attribute__((ext_vector_type(16))) _Float16 v16h;
typedef __attribute__((ext_vector_type(8)))  _Float16 v8h;
typedef __attribute__((ext_vector_type(8)))  float    v8f;
typedef __attribute__((ext_vector_type(4)))  float    v4f;

#define TK 32
#define FRAG_STRIDE 24   // 16 halves used + 8 pad -> 48B lane stride (16B aligned)

// C[M x N] = alpha * (A @ op(B)) + bias
// TRANSB=1: B is N x K row-major (weights) -> C = A @ B^T
// TRANSB=0: B is K x N row-major           -> C = A @ B
// VEC=1 (requires TRANSB=1, K%32==0, lda%4==0, ldb%4==0, 16B-aligned A/B):
//        vectorized float4 staging loads, no k-guards.
template <int TRANSB, int VEC>
__global__ __launch_bounds__(256) void wmma_gemm(
    const float* __restrict__ A, int lda,
    const float* __restrict__ Bm, int ldb,
    const float* __restrict__ bias,
    float* __restrict__ C, int ldc,
    int M, int N, int K, float alpha)
{
    // Fragment-native LDS: frag[16-row tile][lane][j]. Written pre-swizzled so
    // each lane's 16 fragment halves are contiguous (2x ds_load_b128 to read).
    __shared__ __attribute__((aligned(16))) half_t Afrag[4][32][FRAG_STRIDE];
    __shared__ __attribute__((aligned(16))) half_t Bfrag[4][32][FRAG_STRIDE];

    const int tid  = threadIdx.x;
    const int lane = tid & 31;
    const int wave = tid >> 5;

    const int blockM = blockIdx.y * 64;
    const int blockN = blockIdx.x * 64;

    const int wm = wave & 3;          // A fragment tile (16 rows each)
    const int wn = (wave >> 2) * 2;   // B fragment tiles wn, wn+1

    // ---- staging coordinates: thread loads 8 consecutive k of one row ----
    const int sr  = tid >> 2;          // tile row (A m-row / B n-row): 0..63
    const int sc8 = (tid & 3) * 8;     // k chunk start: 0,8,16,24
    const int s_t16 = sr >> 4;
    const int s_m   = sr & 15;
    // A-fragment k-permutation: c8 -> (half lane group, j0)
    const int a_h  = (sc8 >> 3) & 1;
    const int a_j0 = (sc8 >> 4) << 3;
    // B-fragment k-permutation
    const int b_h  = (sc8 >> 3) & 1;
    const int b_j0 = (sc8 >> 4) << 3;
    half_t* a_dst = &Afrag[s_t16][a_h * 16 + s_m][a_j0];
    half_t* b_dst = &Bfrag[s_t16][b_h * 16 + s_m][b_j0];

    const int ga_r  = blockM + sr;
    const int gb_n  = blockN + sr;
    // Row-clamped pointers: OOB rows read valid (clamped) data; the garbage
    // C rows/cols they produce are never stored, so no row masks are needed.
    const int ga_r2 = (ga_r < M) ? ga_r : (M - 1);
    const int gb_n2 = (gb_n < N) ? gb_n : (N - 1);
    const float* __restrict__ Arow = A + (long)ga_r2 * lda;
    const float* __restrict__ Brow = TRANSB ? (Bm + (long)gb_n2 * ldb) : Bm;

    v8f acc0 = {}; v8f acc1 = {};

    for (int k0 = 0; k0 < K; k0 += TK) {
        v8h av, bv;
        if (VEC) {
            const v4f* pa = (const v4f*)(Arow + k0 + sc8);
            v4f a0 = pa[0], a1 = pa[1];
            const v4f* pb = (const v4f*)(Brow + k0 + sc8);
            v4f bb0 = pb[0], bb1 = pb[1];
            #pragma unroll
            for (int i = 0; i < 4; ++i) {
                av[i]     = (half_t)a0[i];
                av[i + 4] = (half_t)a1[i];
                bv[i]     = (half_t)bb0[i];
                bv[i + 4] = (half_t)bb1[i];
            }
        } else {
            // branchless k-guard: multiply-mask keeps loads unconditional
            #pragma unroll
            for (int i = 0; i < 8; ++i) {
                int gk  = k0 + sc8 + i;
                int gk2 = (gk < K) ? gk : (K - 1);
                float km = (gk < K) ? 1.0f : 0.0f;
                float va = Arow[gk2];
                float vb = TRANSB ? Brow[gk2] : Bm[(long)gk2 * ldb + gb_n2];
                av[i] = (half_t)(va * km);
                bv[i] = (half_t)(vb * km);
            }
        }
        if (k0 + TK < K) {
            __builtin_prefetch(&Arow[k0 + TK + sc8], 0, 3);
            if (TRANSB) __builtin_prefetch(&Brow[k0 + TK + sc8], 0, 3);
        }
        *(v8h*)a_dst = av;           // one 16B LDS store each
        *(v8h*)b_dst = bv;
        __syncthreads();

        union F16 { v16h v; v8h h[2]; };
        F16 a, b0, b1;
        a.h[0]  = *(const v8h*)&Afrag[wm][lane][0];
        a.h[1]  = *(const v8h*)&Afrag[wm][lane][8];
        b0.h[0] = *(const v8h*)&Bfrag[wn][lane][0];
        b0.h[1] = *(const v8h*)&Bfrag[wn][lane][8];
        b1.h[0] = *(const v8h*)&Bfrag[wn + 1][lane][0];
        b1.h[1] = *(const v8h*)&Bfrag[wn + 1][lane][8];

        acc0 = __builtin_amdgcn_wmma_f32_16x16x32_f16(false, a.v, false, b0.v, (short)0, acc0, false, false);
        acc1 = __builtin_amdgcn_wmma_f32_16x16x32_f16(false, a.v, false, b1.v, (short)0, acc1, false, false);
        __syncthreads();
    }

    // C/D layout: VGPR i -> M = i (lanes 0-15) / i+8 (lanes 16-31), N = lane&15
    const int halfsel = lane >> 4;
    const int mrow    = lane & 15;
    #pragma unroll
    for (int i = 0; i < 8; ++i) {
        int gm = blockM + wm * 16 + i + halfsel * 8;
        if (gm < M) {
            int gn0 = blockN + wn * 16 + mrow;
            if (gn0 < N) C[(long)gm * ldc + gn0] = alpha * acc0[i] + (bias ? bias[gn0] : 0.0f);
            int gn1 = gn0 + 16;
            if (gn1 < N) C[(long)gm * ldc + gn1] = alpha * acc1[i] + (bias ? bias[gn1] : 0.0f);
        }
    }
}

// ----------------------- GRU gate (one block per row, 256 = H threads) ------
__global__ __launch_bounds__(256) void gru_gate(
    const float* __restrict__ xw, const float* __restrict__ gh,
    float* __restrict__ h,
    float* __restrict__ seq_out, int seq_S, int t,
    float* __restrict__ last_out, const int* __restrict__ len)
{
    int row = blockIdx.x;
    int j = threadIdx.x;
    long base = (long)row * 768;
    float xr = xw[base + j], xz = xw[base + 256 + j], xn = xw[base + 512 + j];
    float hr = gh[base + j], hz = gh[base + 256 + j], hn = gh[base + 512 + j];
    float r = 1.0f / (1.0f + __expf(-(xr + hr)));
    float z = 1.0f / (1.0f + __expf(-(xz + hz)));
    float n = tanhf(xn + r * hn);
    float hold = h[(long)row * 256 + j];
    float hnew = (1.0f - z) * n + z * hold;
    h[(long)row * 256 + j] = hnew;
    if (seq_out) seq_out[((long)row * seq_S + t) * 256 + j] = hnew;
    if (last_out && len && (len[row] - 1 == t)) last_out[(long)row * 256 + j] = hnew;
}

// ----------------------- concat builders ------------------------------------
__global__ void build_mrv(const float* __restrict__ his_last,
                          const float* __restrict__ inter_r,
                          float* __restrict__ M_rv)
{
    long idx = (long)blockIdx.x * blockDim.x + threadIdx.x;
    if (idx >= 6144L * 384) return;
    int row = (int)(idx / 384), c = (int)(idx % 384);
    M_rv[idx] = (c < 256) ? his_last[(long)row * 256 + c]
                          : inter_r[(long)row * 128 + (c - 256)];
}

__global__ void build_mpv(const float* __restrict__ flat_h,
                          const float* __restrict__ intra_x,
                          float* __restrict__ m_pv)
{
    long idx = (long)blockIdx.x * blockDim.x + threadIdx.x;
    if (idx >= 1024L * 383) return;
    int row = (int)(idx / 383), c = (int)(idx % 383);
    m_pv[idx] = (c < 256) ? flat_h[(long)row * 256 + c]
                          : intra_x[(long)row * 128 + (c - 256)];
}

__global__ void build_hp(const float* __restrict__ flat_h,
                         const float* __restrict__ intra_x,
                         float* __restrict__ hp)
{
    long idx = (long)blockIdx.x * blockDim.x + threadIdx.x;
    if (idx >= 1024L * 257) return;
    int row = (int)(idx / 257), c = (int)(idx % 257);
    hp[idx] = (c < 256) ? flat_h[(long)row * 256 + c]
                        : intra_x[(long)row * 128 + 127];
}

// ----------------------- v_v attention (R=6 keys, 2 heads of 128) -----------
__global__ __launch_bounds__(256) void vv_attn(
    const float* __restrict__ q, const float* __restrict__ k,
    const float* __restrict__ v, float* __restrict__ out)
{
    __shared__ float red[256];
    __shared__ float pw[2][6];
    int i = blockIdx.x, tid = threadIdx.x;
    int hd = tid >> 7, dl = tid & 127;
    float qv = q[(long)i * 256 + tid];
    float sc[6];
    for (int r = 0; r < 6; ++r) {
        red[tid] = qv * k[((long)i * 6 + r) * 256 + tid];
        __syncthreads();
        for (int off = 64; off > 0; off >>= 1) {
            if (dl < off) red[tid] += red[tid + off];
            __syncthreads();
        }
        sc[r] = red[hd << 7] * 0.08838834764831845f;  // 1/sqrt(128)
        __syncthreads();
    }
    if (dl == 0) {
        float mx = sc[0];
        for (int r = 1; r < 6; ++r) mx = fmaxf(mx, sc[r]);
        float s = 0.0f, e[6];
        for (int r = 0; r < 6; ++r) { e[r] = __expf(sc[r] - mx); s += e[r]; }
        for (int r = 0; r < 6; ++r) pw[hd][r] = e[r] / s;
    }
    __syncthreads();
    float o = 0.0f;
    for (int r = 0; r < 6; ++r) o += pw[hd][r] * v[((long)i * 6 + r) * 256 + tid];
    out[(long)i * 256 + tid] = o;
}

// ----------------------- causal softmax over 64-wide rows -------------------
__global__ __launch_bounds__(64) void softmax_causal(float* __restrict__ scores)
{
    __shared__ float red[64];
    int rid = blockIdx.x;
    int t = threadIdx.x;
    int qi = rid & 63;
    float s = (t <= qi) ? scores[(long)rid * 64 + t] : -1e30f;
    red[t] = s; __syncthreads();
    for (int off = 32; off > 0; off >>= 1) {
        if (t < off) red[t] = fmaxf(red[t], red[t + off]);
        __syncthreads();
    }
    float mx = red[0]; __syncthreads();
    float e = (t <= qi) ? __expf(s - mx) : 0.0f;
    red[t] = e; __syncthreads();
    for (int off = 32; off > 0; off >>= 1) {
        if (t < off) red[t] += red[t + off];
        __syncthreads();
    }
    scores[(long)rid * 64 + t] = e / red[0];
}

// ----------------------- weighted combine + feature concat ------------------
__global__ void combine_feat(const float* __restrict__ wr,
                             const float* __restrict__ vv,
                             const float* __restrict__ vh,
                             const float* __restrict__ flat_h,
                             const float* __restrict__ intra_x,
                             float* __restrict__ feat)
{
    long idx = (long)blockIdx.x * blockDim.x + threadIdx.x;
    if (idx >= 1024L * 639) return;
    int row = (int)(idx / 639), c = (int)(idx % 639);
    float m = fmaxf(wr[0], wr[1]);
    float e0 = __expf(wr[0] - m), e1 = __expf(wr[1] - m);
    float w0 = e0 / (e0 + e1), w1 = e1 / (e0 + e1);
    float v;
    if (c < 256)      v = w0 * vv[(long)row * 256 + c] + w1 * vh[(long)row * 256 + c];
    else if (c < 512) v = flat_h[(long)row * 256 + (c - 256)];
    else              v = intra_x[(long)row * 128 + (c - 512)];
    feat[idx] = v;
}

// ---------------------------------------------------------------------------
extern "C" void kernel_launch(void* const* d_in, const int* in_sizes, int n_in,
                              void* d_out, int out_size, void* d_ws, size_t ws_size,
                              hipStream_t stream)
{
    (void)in_sizes; (void)n_in; (void)out_size; (void)ws_size;

    const float* intra_x   = (const float*)d_in[0];
    const float* inter_his = (const float*)d_in[1];
    const float* inter_r   = (const float*)d_in[2];
    /* d_in[3] intra_mask: all-ones, unused by reference */
    const int*   inter_len = (const int*)d_in[4];
    const float* w_ih = (const float*)d_in[5];
    const float* w_hh = (const float*)d_in[6];
    const float* b_ih = (const float*)d_in[7];
    const float* b_hh = (const float*)d_in[8];
    const float* iq_w = (const float*)d_in[9];
    const float* iq_b = (const float*)d_in[10];
    const float* ik_w = (const float*)d_in[11];
    const float* ik_b = (const float*)d_in[12];
    const float* iv_w = (const float*)d_in[13];
    const float* iv_b = (const float*)d_in[14];
    const float* io_w = (const float*)d_in[15];
    const float* io_b = (const float*)d_in[16];
    const float* aq_w = (const float*)d_in[17];
    const float* aq_b = (const float*)d_in[18];
    const float* ak_w = (const float*)d_in[19];
    const float* ak_b = (const float*)d_in[20];
    const float* av_w = (const float*)d_in[21];
    const float* av_b = (const float*)d_in[22];
    const float* ao_w = (const float*)d_in[23];
    const float* ao_b = (const float*)d_in[24];
    const float* wr   = (const float*)d_in[25];
    const float* ln_w = (const float*)d_in[26];
    const float* ln_b = (const float*)d_in[27];

    // ---- workspace bump allocation (floats; all offsets multiples of 4) ----
    size_t off = 0;
    float* base = (float*)d_ws;
    auto alloc = [&](size_t n) { float* p = base + off; off += n; return p; };
    float* h_inter  = alloc(6144 * 256);
    float* his_last = alloc(6144 * 256);
    float* flat_h   = alloc(1024 * 256);
    float* h_intra  = alloc(16 * 256);
    float* xw_i     = alloc(16 * 768);
    float* gh_i     = alloc(16 * 768);
    float* xw       = alloc((size_t)6144 * 768);
    float* gh       = alloc((size_t)6144 * 768);
    // aliases inside xw region (free after GRUs complete)
    float* M_rv  = xw;
    float* kv    = M_rv + (size_t)6144 * 384;
    float* qv    = kv + (size_t)6144 * 256;
    float* m_pv  = qv + (size_t)1024 * 256;
    // aliases inside gh region
    float* vv_    = gh;
    float* attn_o = vv_ + (size_t)6144 * 256;
    float* v_v    = attn_o + (size_t)1024 * 256;
    float* qh     = v_v + (size_t)1024 * 256;
    float* kh     = qh + (size_t)1024 * 256;
    float* hp     = kh + (size_t)1024 * 256;
    float* vh_p   = hp + (size_t)1024 * 257 + 3;   // 3-float pad; vh_p and later bufs lose 16B alignment
    float* scores = vh_p + (size_t)1024 * 256;
    float* attn_h = scores + (size_t)32 * 64 * 64;
    float* vh_o   = attn_h + (size_t)1024 * 256;
    float* feat   = vh_o + (size_t)1024 * 256;

    auto gemmT = [&](const float* A, int lda, const float* W, int ldw,
                     const float* bias, float* C, int ldc,
                     int M, int N, int K, float alpha) {
        dim3 g((N + 63) / 64, (M + 63) / 64);
        bool vec = (K % 32 == 0) && (lda % 4 == 0) && (ldw % 4 == 0);
        if (vec)
            wmma_gemm<1, 1><<<g, 256, 0, stream>>>(A, lda, W, ldw, bias, C, ldc, M, N, K, alpha);
        else
            wmma_gemm<1, 0><<<g, 256, 0, stream>>>(A, lda, W, ldw, bias, C, ldc, M, N, K, alpha);
    };
    auto gemmN = [&](const float* A, int lda, const float* Bm, int ldb,
                     const float* bias, float* C, int ldc,
                     int M, int N, int K, float alpha) {
        dim3 g((N + 63) / 64, (M + 63) / 64);
        wmma_gemm<0, 0><<<g, 256, 0, stream>>>(A, lda, Bm, ldb, bias, C, ldc, M, N, K, alpha);
    };

    hipMemsetAsync(h_inter, 0, (size_t)6144 * 256 * sizeof(float), stream);
    hipMemsetAsync(h_intra, 0, (size_t)16 * 256 * sizeof(float), stream);

    // ---- inter GRU (batch 6144, L=24), fused last-hidden gather ----
    for (int t = 0; t < 24; ++t) {
        gemmT(inter_his + (size_t)t * 128, 24 * 128, w_ih, 128, b_ih, xw, 768, 6144, 768, 128, 1.0f);
        gemmT(h_inter, 256, w_hh, 256, b_hh, gh, 768, 6144, 768, 256, 1.0f);
        gru_gate<<<6144, 256, 0, stream>>>(xw, gh, h_inter, nullptr, 0, t, his_last, inter_len);
    }

    // ---- intra GRU (batch 16, S=64), store full hidden sequence ----
    for (int t = 0; t < 64; ++t) {
        gemmT(intra_x + (size_t)t * 128, 64 * 128, w_ih, 128, b_ih, xw_i, 768, 16, 768, 128, 1.0f);
        gemmT(h_intra, 256, w_hh, 256, b_hh, gh_i, 768, 16, 768, 256, 1.0f);
        gru_gate<<<16, 256, 0, stream>>>(xw_i, gh_i, h_intra, flat_h, 64, t, nullptr, nullptr);
    }

    // ---- v_v path: cross attention over R=6 ----
    build_mpv<<<(1024 * 383 + 255) / 256, 256, 0, stream>>>(flat_h, intra_x, m_pv);
    build_mrv<<<(6144 * 384 + 255) / 256, 256, 0, stream>>>(his_last, inter_r, M_rv);
    gemmT(m_pv, 383, iq_w, 383, iq_b, qv, 256, 1024, 256, 383, 1.0f);
    gemmT(M_rv, 384, ik_w, 383, ik_b, kv, 256, 6144, 256, 383, 1.0f);  // M_pv = M_rv[:, :383]
    gemmT(M_rv, 384, iv_w, 384, iv_b, vv_, 256, 6144, 256, 384, 1.0f);
    vv_attn<<<1024, 256, 0, stream>>>(qv, kv, vv_, attn_o);
    gemmT(attn_o, 256, io_w, 256, io_b, v_v, 256, 1024, 256, 256, 1.0f);

    // ---- v_h path: causal self-attention over S=64 ----
    gemmT(intra_x, 128, aq_w, 127, aq_b, qh, 256, 1024, 256, 127, 1.0f);
    gemmT(intra_x, 128, ak_w, 127, ak_b, kh, 256, 1024, 256, 127, 1.0f);
    build_hp<<<(1024 * 257 + 255) / 256, 256, 0, stream>>>(flat_h, intra_x, hp);
    gemmT(hp, 257, av_w, 257, av_b, vh_p, 256, 1024, 256, 257, 1.0f);
    for (int b = 0; b < 16; ++b) {
        for (int hd = 0; hd < 2; ++hd) {
            const float* Qs = qh + (size_t)b * 64 * 256 + hd * 128;
            const float* Ks = kh + (size_t)b * 64 * 256 + hd * 128;
            float* Sc = scores + ((size_t)b * 2 + hd) * 64 * 64;
            gemmT(Qs, 256, Ks, 256, nullptr, Sc, 64, 64, 64, 128, 0.08838834764831845f);
        }
    }
    softmax_causal<<<2048, 64, 0, stream>>>(scores);
    for (int b = 0; b < 16; ++b) {
        for (int hd = 0; hd < 2; ++hd) {
            const float* Ps = scores + ((size_t)b * 2 + hd) * 64 * 64;
            const float* Vs = vh_p + (size_t)b * 64 * 256 + hd * 128;
            float* Os = attn_h + (size_t)b * 64 * 256 + hd * 128;
            gemmN(Ps, 64, Vs, 256, nullptr, Os, 256, 64, 128, 64, 1.0f);
        }
    }
    gemmT(attn_h, 256, ao_w, 256, ao_b, vh_o, 256, 1024, 256, 256, 1.0f);

    // ---- combine + final linear ----
    combine_feat<<<(1024 * 639 + 255) / 256, 256, 0, stream>>>(wr, v_v, vh_o, flat_h, intra_x, feat);
    gemmT(feat, 639, ln_w, 639, ln_b, (float*)d_out, 256, 1024, 256, 639, 1.0f);
}
